// graphcl_44006234915448
// MI455X (gfx1250) — matrix operand order repaired and checked by reading, not compiled, not run on hardware
//
#include <hip/hip_runtime.h>

typedef __attribute__((ext_vector_type(16))) __bf16 v16bf;
typedef __attribute__((ext_vector_type(8)))  __bf16 bf16x8;
typedef __attribute__((ext_vector_type(8)))  float  v8f;

#define B_N   8192
#define D_K   256
#define TILE  128
#define LDST  264   // padded LDS row stride (bf16 elems): 264*2B = 528B = 33*16B -> bank-conflict-free b128 frags

// ---------------------------------------------------------------------------
// Kernel 1: row L2-normalize x1/x2, emit bf16 (RNE) normalized copies; zero row_sum.
// grid = (8192, 2), block = 256 (one row per block, one feature per thread)
// ---------------------------------------------------------------------------
__global__ __launch_bounds__(256) void ntxent_norm(
    const float* __restrict__ x1, const float* __restrict__ x2,
    unsigned short* __restrict__ x1n, unsigned short* __restrict__ x2n,
    float* __restrict__ row_sum, float* __restrict__ diag)
{
    __shared__ float red[256];
    const int t   = threadIdx.x;
    const int row = blockIdx.x;
    const float* __restrict__ src = blockIdx.y ? x2 : x1;
    unsigned short* __restrict__ dst = blockIdx.y ? x2n : x1n;

    float v = src[row * D_K + t];
    red[t] = v * v;
    __syncthreads();
    for (int s = 128; s > 0; s >>= 1) {
        if (t < s) red[t] += red[t + s];
        __syncthreads();
    }
    float nv = v * rsqrtf(red[0]);
    // fp32 -> bf16 round-to-nearest-even
    unsigned int u = __float_as_uint(nv);
    u += 0x7FFFu + ((u >> 16) & 1u);
    dst[row * D_K + t] = (unsigned short)(u >> 16);

    if (blockIdx.y == 0 && t == 0) { row_sum[row] = 0.0f; diag[row] = 1.0f; }
}

// ---------------------------------------------------------------------------
// Kernel 2: fused cos-sim GEMM (bf16 WMMA) + exp + diagonal + row-sum reduce.
// grid = (64 col-tiles, 64 row-tiles), block = 256 = 8 wave32s.
// Each wave: 32 rows x 64 cols = 2x4 tiles of 16x16, K=256 in 8 steps of 32.
// ---------------------------------------------------------------------------
__global__ __launch_bounds__(256) void ntxent_gemm(
    const unsigned short* __restrict__ x1n,
    const unsigned short* __restrict__ x2n,
    float* __restrict__ row_sum, float* __restrict__ diag)
{
    __shared__ unsigned short As[TILE * LDST];   // 67584 B
    __shared__ unsigned short Bs[TILE * LDST];   // 67584 B  (total 132 KB of 320 KB WGP LDS)

    const int t    = threadIdx.x;
    const int row0 = blockIdx.y * TILE;
    const int col0 = blockIdx.x * TILE;

    // Stage both 128x256 bf16 tiles into LDS in 16B chunks (coalesced b128 load + ds_store_b128)
    for (int c = t; c < TILE * (D_K / 8); c += 256) {   // 4096 chunks each
        const int r  = c >> 5;
        const int ch = c & 31;
        *(uint4*)&As[r * LDST + ch * 8] = *(const uint4*)&x1n[(row0 + r) * D_K + ch * 8];
        *(uint4*)&Bs[r * LDST + ch * 8] = *(const uint4*)&x2n[(col0 + r) * D_K + ch * 8];
    }
    __syncthreads();

    const int lane = t & 31;
    const int w    = t >> 5;
    const int wm   = w & 3;      // M sub-block: 32 rows
    const int wn   = w >> 2;     // N sub-block: 64 cols
    const int g    = lane >> 4;  // half-wave group
    const int hl   = lane & 15;

    const v8f vzero = {0.f, 0.f, 0.f, 0.f, 0.f, 0.f, 0.f, 0.f};
    v8f acc[2][4];
#pragma unroll
    for (int mt = 0; mt < 2; ++mt)
#pragma unroll
        for (int nt = 0; nt < 4; ++nt) acc[mt][nt] = vzero;

#pragma unroll
    for (int kk = 0; kk < D_K; kk += 32) {
        // A fragments (16x32 bf16): lane holds row m=hl; K chunks [kk+8g..+7] and [kk+16+8g..+7]
        v16bf a[2];
#pragma unroll
        for (int mt = 0; mt < 2; ++mt) {
            const unsigned short* p = &As[(wm * 32 + mt * 16 + hl) * LDST + kk + 8 * g];
            bf16x8 lo = *(const bf16x8*)p;
            bf16x8 hi = *(const bf16x8*)(p + 16);
#pragma unroll
            for (int e = 0; e < 8; ++e) { a[mt][e] = lo[e]; a[mt][e + 8] = hi[e]; }
        }
#pragma unroll
        for (int nt = 0; nt < 4; ++nt) {
            // B fragment (32x16 bf16): lane holds column n=hl; K chunks [kk+8g..+7] and [kk+16+8g..+7]
            const unsigned short* p = &Bs[(wn * 64 + nt * 16 + hl) * LDST + kk + 8 * g];
            bf16x8 lo = *(const bf16x8*)p;
            bf16x8 hi = *(const bf16x8*)(p + 16);
            v16bf b;
#pragma unroll
            for (int e = 0; e < 8; ++e) { b[e] = lo[e]; b[e + 8] = hi[e]; }
#pragma unroll
            for (int mt = 0; mt < 2; ++mt) {
                acc[mt][nt] = __builtin_amdgcn_wmma_f32_16x16x32_bf16(
                    false, a[mt], false, b, (short)0, acc[mt][nt], false, false);
            }
        }
    }

    // Epilogue: s = exp(cos/T), capture diagonal, reduce rows.
    // C/D layout: VGPR r of lane -> (M = r + 8g, N = hl)
    const float invT = 10.0f;
#pragma unroll
    for (int mt = 0; mt < 2; ++mt) {
        float rp[8];
#pragma unroll
        for (int r = 0; r < 8; ++r) rp[r] = 0.f;
#pragma unroll
        for (int nt = 0; nt < 4; ++nt) {
            const int j = col0 + wn * 64 + nt * 16 + hl;
#pragma unroll
            for (int r = 0; r < 8; ++r) {
                float s = __expf(acc[mt][nt][r] * invT);
                const int i = row0 + wm * 32 + mt * 16 + 8 * g + r;
                if (i == j) diag[i] = s;
                rp[r] += s;
            }
        }
#pragma unroll
        for (int r = 0; r < 8; ++r) {
            float v = rp[r];
            v += __shfl_xor(v, 1);
            v += __shfl_xor(v, 2);
            v += __shfl_xor(v, 4);
            v += __shfl_xor(v, 8);
            if (hl == 0)
                atomicAdd(&row_sum[row0 + wm * 32 + mt * 16 + 8 * g + r], v);
        }
    }
}

// ---------------------------------------------------------------------------
// Kernel 3: loss = mean( log(sum_other) - log(pos) ),  sum_other = row_sum - diag
// ---------------------------------------------------------------------------
__global__ __launch_bounds__(256) void ntxent_final(
    const float* __restrict__ row_sum, const float* __restrict__ diag,
    float* __restrict__ out)
{
    __shared__ float red[256];
    const int t = threadIdx.x;
    float acc = 0.f;
    for (int i = t; i < B_N; i += 256) {
        const float p  = diag[i];
        const float so = row_sum[i] - p;
        acc += logf(so) - logf(p);
    }
    red[t] = acc;
    __syncthreads();
    for (int s = 128; s > 0; s >>= 1) {
        if (t < s) red[t] += red[t + s];
        __syncthreads();
    }
    if (t == 0) out[0] = red[0] / (float)B_N;
}

extern "C" void kernel_launch(void* const* d_in, const int* in_sizes, int n_in,
                              void* d_out, int out_size, void* d_ws, size_t ws_size,
                              hipStream_t stream) {
    (void)in_sizes; (void)n_in; (void)out_size; (void)ws_size;
    const float* x1 = (const float*)d_in[0];
    const float* x2 = (const float*)d_in[1];

    unsigned short* x1n = (unsigned short*)d_ws;               // 8192*256 bf16 = 4 MB
    unsigned short* x2n = x1n + (size_t)B_N * D_K;             // 4 MB
    float* row_sum = (float*)(x2n + (size_t)B_N * D_K);        // 32 KB
    float* diag    = row_sum + B_N;                            // 32 KB
    float* out     = (float*)d_out;

    ntxent_norm <<<dim3(B_N, 2), 256, 0, stream>>>(x1, x2, x1n, x2n, row_sum, diag);
    ntxent_gemm <<<dim3(B_N / TILE, B_N / TILE), 256, 0, stream>>>(x1n, x2n, row_sum, diag);
    ntxent_final<<<1, 256, 0, stream>>>(row_sum, diag, out);
}
